// BasicBlock_84653805404653
// MI455X (gfx1250) — matrix operand order repaired and checked by reading, not compiled, not run on hardware
//
#include <hip/hip_runtime.h>

// ---------------------------------------------------------------------------
// Types for CDNA5 WMMA
// ---------------------------------------------------------------------------
typedef __attribute__((ext_vector_type(16))) _Float16 v16h;
typedef __attribute__((ext_vector_type(8)))  _Float16 v8h;
typedef __attribute__((ext_vector_type(8)))  float    v8f;

#define D_MODEL 1024
#define D_FF    4096
#define BATCH   8
#define SEQ     1024
#define NHEADS  16
#define DKH     64
#define ROWS    (BATCH * SEQ)   // 8192

__device__ __forceinline__ v16h make_frag(v8h lo, v8h hi) {
  v16h r;
#pragma unroll
  for (int i = 0; i < 8; ++i) { r[i] = lo[i]; r[i + 8] = hi[i]; }
  return r;
}

template <int PAT>
__device__ __forceinline__ float swz(float v) {
  return __int_as_float(__builtin_amdgcn_ds_swizzle(__float_as_int(v), PAT));
}
// group-of-32 xor patterns: offset = (xor<<10) | 0x1F
#define SWZ_X1  0x041F
#define SWZ_X2  0x081F
#define SWZ_X4  0x101F
#define SWZ_X8  0x201F
#define SWZ_X16 0x401F

// ---------------------------------------------------------------------------
// CDNA5 async global->LDS copy (GLOBAL_LOAD_ASYNC_TO_LDS_B128, ASYNCcnt).
// LDS operand = low 32 bits of the flat shared address (ISA 10.2).
// ---------------------------------------------------------------------------
__device__ __forceinline__ void async_b128(const _Float16* gsrc, _Float16* ldst) {
  unsigned loff = (unsigned)(size_t)ldst;
  unsigned long long ga = (unsigned long long)(size_t)gsrc;
  asm volatile("global_load_async_to_lds_b128 %0, %1, off"
               :
               : "v"(loff), "v"(ga)
               : "memory");
}
__device__ __forceinline__ void wait_async() {
  asm volatile("s_wait_asynccnt 0x0" ::: "memory");
}

// ---------------------------------------------------------------------------
// fp32 -> fp16 conversion (weights / activations feeding WMMA)
// ---------------------------------------------------------------------------
__global__ void cvt_kernel(const float* __restrict__ in, _Float16* __restrict__ out,
                           size_t n) {
  size_t i = (size_t)blockIdx.x * blockDim.x + threadIdx.x;
  size_t stride = (size_t)gridDim.x * blockDim.x;
  for (; i < n; i += stride) out[i] = (_Float16)in[i];
}

// ---------------------------------------------------------------------------
// WMMA GEMM: C[M,N] = A[M,K](f16) * B[K,N](f16) + bias (+resid) (+relu)
// Block: 128 threads (4 waves). Block tile 64x128, K-step 32, double-buffered
// LDS so the ASYNCcnt DMA of tile t+1 overlaps the 8 WMMAs on tile t, with a
// single barrier per K-step.
// ---------------------------------------------------------------------------
struct B4 { v8h r0, r1, r2, r3; };

__device__ __forceinline__ B4 load_B4(const _Float16* B, int N, int n0, int bkk,
                                      int bcol, int kt) {
  const v8h* s =
      reinterpret_cast<const v8h*>(B + (size_t)(kt + bkk) * N + n0 + bcol);
  B4 r;
  r.r0 = s[0]; r.r1 = s[1]; r.r2 = s[2]; r.r3 = s[3];
  return r;
}
__device__ __forceinline__ void store_BT(const B4& v, _Float16* BsTb, int bkk,
                                         int bcol) {
#pragma unroll
  for (int e = 0; e < 8; ++e) {
    BsTb[(bcol + e) * 40 + bkk]      = v.r0[e];
    BsTb[(bcol + 8 + e) * 40 + bkk]  = v.r1[e];
    BsTb[(bcol + 16 + e) * 40 + bkk] = v.r2[e];
    BsTb[(bcol + 24 + e) * 40 + bkk] = v.r3[e];
  }
}
__device__ __forceinline__ void stage_A_async(const _Float16* A, int K, int m0,
                                              int arow, int acol, int kt,
                                              _Float16* Asb) {
  const _Float16* g = A + (size_t)(m0 + arow) * K + kt + acol;
  _Float16* d = Asb + arow * 40 + acol;
  async_b128(g, d);
  async_b128(g + 8, d + 8);
}

template <bool HAS_RESID, bool HAS_OUTF, bool HAS_OUTH, bool RELU>
__global__ __launch_bounds__(128) void gemm_wmma(
    const _Float16* __restrict__ A, const _Float16* __restrict__ B,
    const float* __restrict__ bias, const float* __restrict__ resid,
    float* __restrict__ outF, _Float16* __restrict__ outH,
    int M, int N, int K) {
  __shared__ _Float16 As[2][64 * 40];
  __shared__ _Float16 BsT[2][128 * 40];

  const int tid = threadIdx.x;
  const int lane = tid & 31;
  const int wid = tid >> 5;
  const int lhalf = lane >> 4;   // 0: lanes 0-15, 1: lanes 16-31
  const int l16 = lane & 15;
  const int m0 = blockIdx.y * 64;
  const int n0 = blockIdx.x * 128;
  const int wm = (wid >> 1) * 32;
  const int wn = (wid & 1) * 64;

  v8f c[2][4] = {};

  const int arow = tid >> 1, acol = (tid & 1) * 16;   // A tile 64x32
  const int bkk = tid >> 2, bcol = (tid & 3) * 32;    // B tile 32x128

  // prologue: stage tile 0 into buffer 0
  stage_A_async(A, K, m0, arow, acol, 0, &As[0][0]);
  {
    B4 b0 = load_B4(B, N, n0, bkk, bcol, 0);
    store_BT(b0, &BsT[0][0], bkk, bcol);
  }
  wait_async();
  __syncthreads();

  int cur = 0;
  for (int kt = 0; kt < K; kt += 32) {
    const _Float16* Asc = &As[cur][0];
    const _Float16* BsTc = &BsT[cur][0];
    const bool has_next = (kt + 32 < K);

    B4 bnext = {};
    if (has_next) {
      bnext = load_B4(B, N, n0, bkk, bcol, kt + 32);          // in flight
      stage_A_async(A, K, m0, arow, acol, kt + 32, &As[cur ^ 1][0]);
      if (kt + 64 < K) {  // global_prefetch_b8 for tile t+2
        __builtin_prefetch(A + (size_t)(m0 + arow) * K + kt + 64 + acol, 0, 1);
        __builtin_prefetch(B + (size_t)(kt + 64 + bkk) * N + n0 + bcol, 0, 1);
      }
    }

    // fragments from current buffer
    v16h af[2], bf[4];
#pragma unroll
    for (int i = 0; i < 2; ++i) {
      int row = wm + 16 * i + l16;
      int cb = lhalf * 8;  // A frag: a[0..7] k=cb.., a[8..15] k=cb+16..
      af[i] = make_frag(*reinterpret_cast<const v8h*>(&Asc[row * 40 + cb]),
                        *reinterpret_cast<const v8h*>(&Asc[row * 40 + cb + 16]));
    }
#pragma unroll
    for (int j = 0; j < 4; ++j) {
      int nn = wn + 16 * j + l16;
      int cb = lhalf * 8;  // B frag: b[0..7] k=cb.., b[8..15] k=cb+16..
      bf[j] = make_frag(*reinterpret_cast<const v8h*>(&BsTc[nn * 40 + cb]),
                        *reinterpret_cast<const v8h*>(&BsTc[nn * 40 + cb + 16]));
    }
#pragma unroll
    for (int i = 0; i < 2; ++i)
#pragma unroll
      for (int j = 0; j < 4; ++j)
        c[i][j] = __builtin_amdgcn_wmma_f32_16x16x32_f16(
            false, af[i], false, bf[j], (short)0, c[i][j], false, false);

    if (has_next) store_BT(bnext, &BsT[cur ^ 1][0], bkk, bcol);
    wait_async();
    __syncthreads();
    cur ^= 1;
  }

  // epilogue: C layout lane holds rows r+lhalf*8 (r=0..7), col l16
#pragma unroll
  for (int i = 0; i < 2; ++i)
#pragma unroll
    for (int j = 0; j < 4; ++j) {
      int col = n0 + wn + 16 * j + l16;
      float bv = bias[col];
#pragma unroll
      for (int r = 0; r < 8; ++r) {
        int row = m0 + wm + 16 * i + r + lhalf * 8;
        size_t idx = (size_t)row * N + col;
        float v = c[i][j][r] + bv;
        if constexpr (HAS_RESID) v += resid[idx];
        if constexpr (RELU) v = fmaxf(v, 0.0f);
        if constexpr (HAS_OUTF) outF[idx] = v;
        if constexpr (HAS_OUTH) outH[idx] = (_Float16)v;
      }
    }
}

// ---------------------------------------------------------------------------
// Flash attention, causal (j < i), first query row zeroed (reference zero_pad).
// Block: 128 threads = 4 waves; each wave owns one 16-query tile.
// K staged [key][d] via async DMA (stride 72); V staged transposed [d][key]
// (stride 40); P bounced through per-wave LDS to convert C-layout -> A-layout.
// ---------------------------------------------------------------------------
__global__ __launch_bounds__(128) void flash_attn(
    const _Float16* __restrict__ Qh, const _Float16* __restrict__ Kh,
    const _Float16* __restrict__ Vh, _Float16* __restrict__ Oh) {
  __shared__ _Float16 Ks[32 * 72];
  __shared__ _Float16 VsT[64 * 40];
  __shared__ _Float16 Pl[4 * 16 * 40];

  const int tid = threadIdx.x;
  const int lane = tid & 31;
  const int wid = tid >> 5;
  const int lhalf = lane >> 4;
  const int l16 = lane & 15;
  const int bh = blockIdx.y;
  const int b = bh >> 4;
  const int h = bh & 15;
  const int qb = blockIdx.x * 64 + wid * 16;

  const size_t headoff = ((size_t)b * SEQ) * D_MODEL + (size_t)h * DKH;
  const _Float16* Qp = Qh + headoff;
  const _Float16* Kp = Kh + headoff;
  const _Float16* Vp = Vh + headoff;

  // Q fragments (persist in registers): two 16x32 A-frags over d
  v16h qa[2];
#pragma unroll
  for (int dh = 0; dh < 2; ++dh) {
    const _Float16* rp = Qp + (size_t)(qb + l16) * D_MODEL + dh * 32 + lhalf * 8;
    qa[dh] = make_frag(*reinterpret_cast<const v8h*>(rp),
                       *reinterpret_cast<const v8h*>(rp + 16));
  }

  float mst[8], lst[8];
  v8f ctx[4] = {};
#pragma unroll
  for (int r = 0; r < 8; ++r) { mst[r] = -3.0e38f; lst[r] = 0.0f; }

  _Float16* Pw = &Pl[wid * 16 * 40];
  const int nch = blockIdx.x * 2 + 2;  // 32-key chunks covering the causal span
  const int kk = tid >> 2, c0 = (tid & 3) * 16;

  for (int ch = 0; ch < nch; ++ch) {
    const int kb = ch * 32;
    __syncthreads();
    {  // K chunk 32x64 via async DMA, natural layout
      const _Float16* gk = Kp + (size_t)(kb + kk) * D_MODEL + c0;
      _Float16* dk = &Ks[kk * 72 + c0];
      async_b128(gk, dk);
      async_b128(gk + 8, dk + 8);
      // V chunk 32x64, transposed -> VsT[d][key]
      const v8h* sv =
          reinterpret_cast<const v8h*>(Vp + (size_t)(kb + kk) * D_MODEL + c0);
      v8h y0 = sv[0], y1 = sv[1];
#pragma unroll
      for (int e = 0; e < 8; ++e) {
        VsT[(c0 + e) * 40 + kk] = y0[e];
        VsT[(c0 + 8 + e) * 40 + kk] = y1[e];
      }
    }
    if (ch + 1 < nch) {
      __builtin_prefetch(Kp + (size_t)(kb + 32 + kk) * D_MODEL + c0, 0, 1);
      __builtin_prefetch(Vp + (size_t)(kb + 32 + kk) * D_MODEL + c0, 0, 1);
    }
    wait_async();
    __syncthreads();

    // scores = Q K^T : load all 4 K fragments first, then 4 WMMAs
    v16h kf[2][2];
#pragma unroll
    for (int kh2 = 0; kh2 < 2; ++kh2)
#pragma unroll
      for (int dh = 0; dh < 2; ++dh) {
        const _Float16* rp = &Ks[(kh2 * 16 + l16) * 72 + dh * 32 + lhalf * 8];
        kf[kh2][dh] = make_frag(*reinterpret_cast<const v8h*>(rp),
                                *reinterpret_cast<const v8h*>(rp + 16));
      }
    v8f sc[2];
#pragma unroll
    for (int kh2 = 0; kh2 < 2; ++kh2) {
      v8f acc = {};
      acc = __builtin_amdgcn_wmma_f32_16x16x32_f16(false, qa[0], false,
                                                   kf[kh2][0], (short)0, acc,
                                                   false, false);
      acc = __builtin_amdgcn_wmma_f32_16x16x32_f16(false, qa[1], false,
                                                   kf[kh2][1], (short)0, acc,
                                                   false, false);
      sc[kh2] = acc;
    }

    // scale + causal mask (j < i)
#pragma unroll
    for (int kh2 = 0; kh2 < 2; ++kh2)
#pragma unroll
      for (int r = 0; r < 8; ++r) {
        int q = qb + r + lhalf * 8;
        int kcol = kb + kh2 * 16 + l16;
        float v = sc[kh2][r] * 0.125f;
        sc[kh2][r] = (kcol < q) ? v : -1.0e32f;
      }

    // online softmax per row (rows live in 16-lane halves)
#pragma unroll
    for (int r = 0; r < 8; ++r) {
      float rm = fmaxf(sc[0][r], sc[1][r]);
      rm = fmaxf(rm, swz<SWZ_X1>(rm));
      rm = fmaxf(rm, swz<SWZ_X2>(rm));
      rm = fmaxf(rm, swz<SWZ_X4>(rm));
      rm = fmaxf(rm, swz<SWZ_X8>(rm));
      float mnew = fmaxf(mst[r], rm);
      float scale = __expf(mst[r] - mnew);
      float p0 = __expf(sc[0][r] - mnew);
      float p1 = __expf(sc[1][r] - mnew);
      float rs = p0 + p1;
      rs += swz<SWZ_X1>(rs);
      rs += swz<SWZ_X2>(rs);
      rs += swz<SWZ_X4>(rs);
      rs += swz<SWZ_X8>(rs);
      lst[r] = lst[r] * scale + rs;
      mst[r] = mnew;
#pragma unroll
      for (int f = 0; f < 4; ++f) ctx[f][r] *= scale;
      int prow = r + lhalf * 8;
      Pw[prow * 40 + l16] = (_Float16)p0;
      Pw[prow * 40 + 16 + l16] = (_Float16)p1;
    }
    __syncthreads();

    // ctx += P(16x32) @ V(32x64): load P frag + all 4 V frags, then 4 WMMAs
    {
      const _Float16* rp = &Pw[l16 * 40 + lhalf * 8];
      v16h pa = make_frag(*reinterpret_cast<const v8h*>(rp),
                          *reinterpret_cast<const v8h*>(rp + 16));
      v16h vb[4];
#pragma unroll
      for (int f = 0; f < 4; ++f) {
        const _Float16* vp = &VsT[(f * 16 + l16) * 40 + lhalf * 8];
        vb[f] = make_frag(*reinterpret_cast<const v8h*>(vp),
                          *reinterpret_cast<const v8h*>(vp + 16));
      }
#pragma unroll
      for (int f = 0; f < 4; ++f)
        ctx[f] = __builtin_amdgcn_wmma_f32_16x16x32_f16(false, pa, false, vb[f],
                                                        (short)0, ctx[f], false,
                                                        false);
    }
  }

  // normalize, zero query row 0 (reference zero_pad), write head slice of concat
#pragma unroll
  for (int f = 0; f < 4; ++f)
#pragma unroll
    for (int r = 0; r < 8; ++r) {
      int q = qb + r + lhalf * 8;
      float v = ctx[f][r] / lst[r];
      if (q == 0) v = 0.0f;
      Oh[((size_t)b * SEQ + q) * D_MODEL + h * DKH + f * 16 + l16] = (_Float16)v;
    }
}

// ---------------------------------------------------------------------------
// LayerNorm over last dim (1024), one block per row.
// ---------------------------------------------------------------------------
template <bool HAS_OUTH>
__global__ __launch_bounds__(256) void ln_kernel(
    const float* __restrict__ in, const float* __restrict__ g,
    const float* __restrict__ bta, float* __restrict__ outF,
    _Float16* __restrict__ outH) {
  const int row = blockIdx.x;
  const float* x = in + (size_t)row * D_MODEL;
  const int lane = threadIdx.x & 31;
  const int wid = threadIdx.x >> 5;

  float s = 0.0f, s2 = 0.0f;
  for (int i = threadIdx.x; i < D_MODEL; i += 256) {
    float v = x[i];
    s += v;
    s2 += v * v;
  }
  s += swz<SWZ_X1>(s);  s2 += swz<SWZ_X1>(s2);
  s += swz<SWZ_X2>(s);  s2 += swz<SWZ_X2>(s2);
  s += swz<SWZ_X4>(s);  s2 += swz<SWZ_X4>(s2);
  s += swz<SWZ_X8>(s);  s2 += swz<SWZ_X8>(s2);
  s += swz<SWZ_X16>(s); s2 += swz<SWZ_X16>(s2);

  __shared__ float sm[8], sq[8];
  if (lane == 0) { sm[wid] = s; sq[wid] = s2; }
  __syncthreads();
  __shared__ float stats[2];
  if (threadIdx.x == 0) {
    float a = 0.0f, b2 = 0.0f;
#pragma unroll
    for (int i = 0; i < 8; ++i) { a += sm[i]; b2 += sq[i]; }
    float mean = a * (1.0f / D_MODEL);
    float var = b2 * (1.0f / D_MODEL) - mean * mean;
    stats[0] = mean;
    stats[1] = rsqrtf(var + 1e-5f);
  }
  __syncthreads();
  float mean = stats[0], inv = stats[1];
  for (int i = threadIdx.x; i < D_MODEL; i += 256) {
    float v = (x[i] - mean) * inv * g[i] + bta[i];
    size_t idx = (size_t)row * D_MODEL + i;
    outF[idx] = v;
    if constexpr (HAS_OUTH) outH[idx] = (_Float16)v;
  }
}

// ---------------------------------------------------------------------------
// Host orchestration
// ---------------------------------------------------------------------------
extern "C" void kernel_launch(void* const* d_in, const int* in_sizes, int n_in,
                              void* d_out, int out_size, void* d_ws,
                              size_t ws_size, hipStream_t stream) {
  (void)in_sizes; (void)n_in; (void)out_size; (void)ws_size;
  const float* x    = (const float*)d_in[0];
  const float* Wq   = (const float*)d_in[1];
  const float* bq   = (const float*)d_in[2];
  const float* Wk   = (const float*)d_in[3];
  const float* bk   = (const float*)d_in[4];
  const float* Wv   = (const float*)d_in[5];
  const float* bv   = (const float*)d_in[6];
  const float* Wo   = (const float*)d_in[7];
  const float* bo   = (const float*)d_in[8];
  const float* W1   = (const float*)d_in[9];
  const float* b1   = (const float*)d_in[10];
  const float* W2   = (const float*)d_in[11];
  const float* b2   = (const float*)d_in[12];
  const float* g1   = (const float*)d_in[13];
  const float* bt1  = (const float*)d_in[14];
  const float* g3   = (const float*)d_in[15];
  const float* bt3  = (const float*)d_in[16];
  float* out = (float*)d_out;

  // carve workspace
  char* p = (char*)d_ws;
  auto carve = [&](size_t bytes) -> char* {
    char* r = p;
    p += (bytes + 255) & ~(size_t)255;
    return r;
  };
  const size_t NE = (size_t)ROWS * D_MODEL;  // 8192*1024
  _Float16* x_h    = (_Float16*)carve(NE * 2);
  _Float16* wq_h   = (_Float16*)carve((size_t)D_MODEL * D_MODEL * 2);
  _Float16* wk_h   = (_Float16*)carve((size_t)D_MODEL * D_MODEL * 2);
  _Float16* wv_h   = (_Float16*)carve((size_t)D_MODEL * D_MODEL * 2);
  _Float16* wo_h   = (_Float16*)carve((size_t)D_MODEL * D_MODEL * 2);
  _Float16* w1_h   = (_Float16*)carve((size_t)D_MODEL * D_FF * 2);
  _Float16* w2_h   = (_Float16*)carve((size_t)D_FF * D_MODEL * 2);
  _Float16* q_h    = (_Float16*)carve(NE * 2);
  _Float16* k_h    = (_Float16*)carve(NE * 2);
  _Float16* v_h    = (_Float16*)carve(NE * 2);
  _Float16* cat_h  = (_Float16*)carve(NE * 2);
  float*    res1   = (float*)carve(NE * 4);
  float*    h1     = (float*)carve(NE * 4);
  _Float16* h1_h   = (_Float16*)carve(NE * 2);
  _Float16* ffn1_h = (_Float16*)carve((size_t)ROWS * D_FF * 2);
  float*    res2   = (float*)carve(NE * 4);

  auto cvt = [&](const float* src, _Float16* dst, size_t n) {
    cvt_kernel<<<2048, 256, 0, stream>>>(src, dst, n);
  };
  cvt(x, x_h, NE);
  cvt(Wq, wq_h, (size_t)D_MODEL * D_MODEL);
  cvt(Wk, wk_h, (size_t)D_MODEL * D_MODEL);
  cvt(Wv, wv_h, (size_t)D_MODEL * D_MODEL);
  cvt(Wo, wo_h, (size_t)D_MODEL * D_MODEL);
  cvt(W1, w1_h, (size_t)D_MODEL * D_FF);
  cvt(W2, w2_h, (size_t)D_FF * D_MODEL);

  // QKV projections (f16 out, consumed by attention)
  dim3 gQKV(D_MODEL / 128, ROWS / 64);
  gemm_wmma<false, false, true, false><<<gQKV, 128, 0, stream>>>(
      x_h, wq_h, bq, nullptr, nullptr, q_h, ROWS, D_MODEL, D_MODEL);
  gemm_wmma<false, false, true, false><<<gQKV, 128, 0, stream>>>(
      x_h, wk_h, bk, nullptr, nullptr, k_h, ROWS, D_MODEL, D_MODEL);
  gemm_wmma<false, false, true, false><<<gQKV, 128, 0, stream>>>(
      x_h, wv_h, bv, nullptr, nullptr, v_h, ROWS, D_MODEL, D_MODEL);

  // causal flash attention -> concat (f16)
  dim3 gAtt(SEQ / 64, BATCH * NHEADS);
  flash_attn<<<gAtt, 128, 0, stream>>>(q_h, k_h, v_h, cat_h);

  // output projection + residual(x) -> res1 (f32)
  gemm_wmma<true, true, false, false><<<gQKV, 128, 0, stream>>>(
      cat_h, wo_h, bo, x, res1, nullptr, ROWS, D_MODEL, D_MODEL);

  // LN1 -> h1 (f32) + h1_h (f16)
  ln_kernel<true><<<ROWS, 256, 0, stream>>>(res1, g1, bt1, h1, h1_h);

  // FFN up + relu -> ffn1_h (f16)
  dim3 gF1(D_FF / 128, ROWS / 64);
  gemm_wmma<false, false, true, true><<<gF1, 128, 0, stream>>>(
      h1_h, w1_h, b1, nullptr, nullptr, ffn1_h, ROWS, D_FF, D_MODEL);

  // FFN down + residual(h1) -> res2 (f32)
  gemm_wmma<true, true, false, false><<<gQKV, 128, 0, stream>>>(
      ffn1_h, w2_h, b2, h1, res2, nullptr, ROWS, D_MODEL, D_FF);

  // LN2 -> final output
  ln_kernel<false><<<ROWS, 256, 0, stream>>>(res2, g3, bt3, out, nullptr);
}
